// CausalSelfAttention_85186381348913
// MI455X (gfx1250) — matrix operand order repaired and checked by reading, not compiled, not run on hardware
//
#include <hip/hip_runtime.h>
#include <hip/hip_bf16.h>

// ---------------- problem constants ----------------
#define BB 2
#define TT 2048
#define DD 768
#define HH 12
#define HD 64
#define NQKV (3 * DD)        // 2304
#define MROWS (BB * TT)      // 4096

// sched_group_barrier masks
#define SG_WMMA  0x0008      // MFMA / WMMA
#define SG_VMRD  0x0020      // VMEM read

typedef _Float16 v16h __attribute__((ext_vector_type(16)));
typedef _Float16 v8h  __attribute__((ext_vector_type(8)));
typedef float    v8f  __attribute__((ext_vector_type(8)));

// ---------------- WMMA helpers ----------------
__device__ __forceinline__ v8f wmma_f16(v16h a, v16h b, v8f c) {
    // D = A(16x32 f16) * B(32x16 f16) + C(16x16 f32)
    return __builtin_amdgcn_wmma_f32_16x16x32_f16(
        /*neg_a=*/false, a, /*neg_b=*/false, b,
        /*c_mod=*/(short)0, c, /*reuse_a=*/false, /*reuse_b=*/false);
}

// A fragment (16x32 f16), A row-major with row stride lda (in halves).
// Layout: lanes 0-15 -> M=lane, halves = K[0..15];
// lanes 16-31 -> M=lane-16, halves = K[16..31].
__device__ __forceinline__ v16h load_a_frag(const _Float16* A, int lda) {
    int lane = threadIdx.x & 31;
    int m    = lane & 15;
    int kb   = (lane >> 4) << 4;            // 0 or 16
    const _Float16* row = A + m * lda;
    v8h lo = *(const v8h*)(row + kb);
    v8h hi = *(const v8h*)(row + kb + 8);
    return __builtin_shufflevector(lo, hi, 0,1,2,3,4,5,6,7,8,9,10,11,12,13,14,15);
}

// B fragment (32x16 f16) with B supplied TRANSPOSED: Bt[n][k], row stride ldb.
// Layout: lanes 0-15 -> N=lane, halves = K[0..15]; lanes 16-31 -> N=lane-16,
// halves = K[16..31].  -> one contiguous 32B load per lane.
__device__ __forceinline__ v16h load_b_frag(const _Float16* Bt, int ldb) {
    int lane = threadIdx.x & 31;
    int n    = lane & 15;
    int kb   = (lane >> 4) << 4;            // 0 or 16
    return *(const v16h*)(Bt + n * ldb + kb);
}

// ---------------- conversion kernels ----------------
__global__ void cvt_f32_to_f16(const float* __restrict__ in,
                               _Float16* __restrict__ out, int n) {
    int i = blockIdx.x * blockDim.x + threadIdx.x;
    if (i < n) out[i] = (_Float16)in[i];
}

// out[n][k] = (f16) in[k][n]   (in is K x N row-major)
__global__ void transpose_f32_to_f16(const float* __restrict__ in,
                                     _Float16* __restrict__ out, int K, int N) {
    int i = blockIdx.x * blockDim.x + threadIdx.x;
    if (i >= K * N) return;
    int k = i / N, n = i % N;
    out[n * K + k] = (_Float16)in[i];
}

// ---------------- GEMM 1: qkv = x @ w_qkv + b_qkv ----------------
// A = xh [4096][768] f16, B = wqkvt [2304][768] f16 (transposed weights).
// Each wave computes a 16(M) x 64(N) tile (4 WMMA accumulators).  The k-loop
// is register double-buffered and the scheduler is pinned via
// sched_group_barrier to [10 VMEM reads] -> [4 WMMA] per iteration, so the
// WMMAs wait with ~10 loads in flight instead of s_wait_loadcnt 0.
// Epilogue: Q (col<768)  -> qk[t][col]   *0.125 (folded softmax scale)
//           K (768..1535)-> qk[t][col]
//           V (>=1536)   -> vt[(b*H+h)*64 + d][t]   (transposed, b128 store)
__global__ void gemm_qkv_kernel(const _Float16* __restrict__ xh,
                                const _Float16* __restrict__ wqkvt,
                                const float* __restrict__ bias,
                                _Float16* __restrict__ qk,
                                _Float16* __restrict__ vt) {
    int wave  = blockIdx.x * (blockDim.x >> 5) + (threadIdx.x >> 5);
    int mtile = wave / (NQKV / 64);          // 0..255
    int ng    = wave % (NQKV / 64);          // 0..35
    int row0  = mtile * 16;

    const _Float16* abase = xh + row0 * DD;
    const _Float16* bbase = wqkvt + (ng * 64) * DD;

    v8f acc[4];
    #pragma unroll
    for (int j = 0; j < 4; ++j) acc[j] = (v8f)(0.0f);

    // prime the pipeline
    v16h a_cur = load_a_frag(abase, DD);
    v16h b_cur[4];
    #pragma unroll
    for (int j = 0; j < 4; ++j) b_cur[j] = load_b_frag(bbase + (j * 16) * DD, DD);

    for (int kk = 0; kk < DD; kk += 32) {
        int kn = (kk + 32 < DD) ? kk + 32 : 0;   // clamped (in-bounds) on last iter
        v16h a_nxt = load_a_frag(abase + kn, DD);
        v16h b_nxt[4];
        #pragma unroll
        for (int j = 0; j < 4; ++j)
            b_nxt[j] = load_b_frag(bbase + (j * 16) * DD + kn, DD);

        #pragma unroll
        for (int j = 0; j < 4; ++j) acc[j] = wmma_f16(a_cur, b_cur[j], acc[j]);

        a_cur = a_nxt;
        #pragma unroll
        for (int j = 0; j < 4; ++j) b_cur[j] = b_nxt[j];

        // pipeline: issue all next-step loads before this step's WMMAs
        __builtin_amdgcn_sched_group_barrier(SG_VMRD, 10, 0);
        __builtin_amdgcn_sched_group_barrier(SG_WMMA, 4, 0);
    }

    int lane = threadIdx.x & 31;
    int lg   = lane >> 4;                    // lane group: rows r + 8*lg
    int nn   = lane & 15;
    int bidx = row0 >> 11;                   // batch (tile never straddles batches)
    int t0   = row0 & (TT - 1);

    #pragma unroll
    for (int j = 0; j < 4; ++j) {
        int col = ng * 64 + j * 16 + nn;
        float bs = bias[col];
        if (col < 2 * DD) {                  // Q or K -> row-major [t][col]
            float scl = (col < DD) ? 0.125f : 1.0f;
            #pragma unroll
            for (int r = 0; r < 8; ++r) {
                int t = row0 + r + 8 * lg;
                qk[t * (2 * DD) + col] = (_Float16)((acc[j][r] + bs) * scl);
            }
        } else {                             // V -> transposed [bh*64+d][t]
            int d    = col - 2 * DD;
            int head = d >> 6;
            int dd   = d & 63;
            v8h tmp;
            #pragma unroll
            for (int r = 0; r < 8; ++r) tmp[r] = (_Float16)(acc[j][r] + bs);
            int base = ((bidx * HH + head) * HD + dd) * TT + t0 + 8 * lg;
            *(v8h*)(vt + base) = tmp;
        }
    }
}

// ---------------- Flash attention ----------------
// One wave per 16-query tile per (b, h).  Online softmax; P transposed
// C-layout -> A-layout through per-wave LDS; PV via transposed-V fragments.
// Pipelined: [16 VMEM reads (V frags + next K frags)] are pinned before the
// [4 score WMMAs], so all fragment loads fly under the softmax + LDS work.
__global__ void flash_attn_kernel(const _Float16* __restrict__ qk,
                                  const _Float16* __restrict__ vt,
                                  _Float16* __restrict__ yh) {
    __shared__ __align__(32) _Float16 ldsP[4][16 * 32];

    int widx  = threadIdx.x >> 5;
    int gwave = blockIdx.x * 4 + widx;       // 0..3071
    int qtile = gwave & 127;
    int bh    = gwave >> 7;                  // 0..23
    int b     = bh / HH;
    int h     = bh % HH;
    int q0    = qtile * 16;

    int lane = threadIdx.x & 31;
    int lg   = lane >> 4;
    int nn   = lane & 15;

    _Float16* P = &ldsP[widx][0];

    // Q fragments (constant for the whole k loop); Q pre-scaled by 1/8.
    const _Float16* qbase = qk + (b * TT + q0) * (2 * DD) + h * HD;
    v16h aq0 = load_a_frag(qbase +  0, 2 * DD);
    v16h aq1 = load_a_frag(qbase + 32, 2 * DD);

    v8f y[4];
    #pragma unroll
    for (int j = 0; j < 4; ++j) y[j] = (v8f)(0.0f);
    float m[8], l[8];
    #pragma unroll
    for (int r = 0; r < 8; ++r) { m[r] = -1.0e30f; l[r] = 0.0f; }

    const _Float16* kbase  = qk + b * TT * (2 * DD) + DD + h * HD;
    const _Float16* vtbase = vt + (bh * HD) * TT;

    int nblk = (q0 + 16 + 31) >> 5;          // causal: need k <= q0+15

    // prime: K fragments for block 0
    // kf[0]: cols k0..k0+15 / d 0..31   kf[1]: cols k0..k0+15 / d 32..63
    // kf[2]: cols +16 / d 0..31         kf[3]: cols +16 / d 32..63
    v16h kf[4];
    kf[0] = load_b_frag(kbase +  0 * (2 * DD) +  0, 2 * DD);
    kf[1] = load_b_frag(kbase +  0 * (2 * DD) + 32, 2 * DD);
    kf[2] = load_b_frag(kbase + 16 * (2 * DD) +  0, 2 * DD);
    kf[3] = load_b_frag(kbase + 16 * (2 * DD) + 32, 2 * DD);

    for (int blk = 0; blk < nblk; ++blk) {
        int k0 = blk * 32;

        // ---- V fragments for this block: issue early, consumed at the end ----
        v16h vf[4];
        #pragma unroll
        for (int j = 0; j < 4; ++j)
            vf[j] = load_b_frag(vtbase + (j * 16) * TT + k0, TT);

        // ---- scores: two 16x16 tiles (cols k0..+15, k0+16..+31) ----
        v8f s0 = (v8f)(0.0f), s1 = (v8f)(0.0f);
        s0 = wmma_f16(aq0, kf[0], s0);
        s0 = wmma_f16(aq1, kf[1], s0);
        s1 = wmma_f16(aq0, kf[2], s1);
        s1 = wmma_f16(aq1, kf[3], s1);

        // ---- issue next block's K fragments (fresh vregs pre-RA: hoistable) ----
        {
            int k1 = (blk + 1 < nblk) ? k0 + 32 : 0;  // clamped, discarded if last
            kf[0] = load_b_frag(kbase + (k1     ) * (2 * DD) +  0, 2 * DD);
            kf[1] = load_b_frag(kbase + (k1     ) * (2 * DD) + 32, 2 * DD);
            kf[2] = load_b_frag(kbase + (k1 + 16) * (2 * DD) +  0, 2 * DD);
            kf[3] = load_b_frag(kbase + (k1 + 16) * (2 * DD) + 32, 2 * DD);
        }

        // ---- causal mask + online softmax update ----
        int c0 = k0 + nn;                    // this lane's column in s0
        float alpha[8];
        #pragma unroll
        for (int r = 0; r < 8; ++r) {
            int row = q0 + r + 8 * lg;
            float v0 = (c0      <= row) ? s0[r] : -1.0e30f;
            float v1 = (c0 + 16 <= row) ? s1[r] : -1.0e30f;
            float mx = fmaxf(v0, v1);
            mx = fmaxf(mx, __shfl_xor(mx, 1));
            mx = fmaxf(mx, __shfl_xor(mx, 2));
            mx = fmaxf(mx, __shfl_xor(mx, 4));
            mx = fmaxf(mx, __shfl_xor(mx, 8));
            float mn = fmaxf(m[r], mx);
            alpha[r] = __expf(m[r] - mn);
            float p0 = __expf(v0 - mn);
            float p1 = __expf(v1 - mn);
            float ps = p0 + p1;
            ps += __shfl_xor(ps, 1);
            ps += __shfl_xor(ps, 2);
            ps += __shfl_xor(ps, 4);
            ps += __shfl_xor(ps, 8);
            l[r] = l[r] * alpha[r] + ps;
            m[r] = mn;
            s0[r] = p0;
            s1[r] = p1;
        }
        #pragma unroll
        for (int j = 0; j < 4; ++j)
            #pragma unroll
            for (int r = 0; r < 8; ++r) y[j][r] *= alpha[r];

        // pipeline pin (same scheduling region: before the wave_barrier):
        // all 16 fragment loads issue before the 4 score WMMAs.
        __builtin_amdgcn_sched_group_barrier(SG_VMRD, 16, 0);
        __builtin_amdgcn_sched_group_barrier(SG_WMMA, 4, 0);

        // ---- P: C-layout -> LDS (16x32 f16, row-major) ----
        __builtin_amdgcn_wave_barrier();     // WAR vs previous iter's reads
        #pragma unroll
        for (int r = 0; r < 8; ++r) {
            P[(r + 8 * lg) * 32 + nn]      = (_Float16)s0[r];
            P[(r + 8 * lg) * 32 + 16 + nn] = (_Float16)s1[r];
        }
        __builtin_amdgcn_wave_barrier();     // RAW: DS unit is in-order

        // ---- y += P @ V  (V stored transposed -> contiguous B frags) ----
        v16h ap = load_a_frag(P, 32);
        #pragma unroll
        for (int j = 0; j < 4; ++j) y[j] = wmma_f16(ap, vf[j], y[j]);
    }

    // ---- normalize + store Y in merged-head [B,T,D] f16 layout ----
    float inv[8];
    #pragma unroll
    for (int r = 0; r < 8; ++r) inv[r] = 1.0f / l[r];
    #pragma unroll
    for (int j = 0; j < 4; ++j) {
        #pragma unroll
        for (int r = 0; r < 8; ++r) {
            int t = q0 + r + 8 * lg;
            yh[(b * TT + t) * DD + h * HD + j * 16 + nn] =
                (_Float16)(y[j][r] * inv[r]);
        }
    }
}

// ---------------- GEMM 3: out = y @ w_out + b_out (fp32 out) ----------------
__global__ void gemm_out_kernel(const _Float16* __restrict__ yh,
                                const _Float16* __restrict__ wot,
                                const float* __restrict__ bias,
                                float* __restrict__ out) {
    int wave  = blockIdx.x * (blockDim.x >> 5) + (threadIdx.x >> 5);
    int mtile = wave / (DD / 64);            // 0..255
    int ng    = wave % (DD / 64);            // 0..11
    int row0  = mtile * 16;

    const _Float16* abase = yh + row0 * DD;
    const _Float16* bbase = wot + (ng * 64) * DD;

    v8f acc[4];
    #pragma unroll
    for (int j = 0; j < 4; ++j) acc[j] = (v8f)(0.0f);

    v16h a_cur = load_a_frag(abase, DD);
    v16h b_cur[4];
    #pragma unroll
    for (int j = 0; j < 4; ++j) b_cur[j] = load_b_frag(bbase + (j * 16) * DD, DD);

    for (int kk = 0; kk < DD; kk += 32) {
        int kn = (kk + 32 < DD) ? kk + 32 : 0;
        v16h a_nxt = load_a_frag(abase + kn, DD);
        v16h b_nxt[4];
        #pragma unroll
        for (int j = 0; j < 4; ++j)
            b_nxt[j] = load_b_frag(bbase + (j * 16) * DD + kn, DD);

        #pragma unroll
        for (int j = 0; j < 4; ++j) acc[j] = wmma_f16(a_cur, b_cur[j], acc[j]);

        a_cur = a_nxt;
        #pragma unroll
        for (int j = 0; j < 4; ++j) b_cur[j] = b_nxt[j];

        __builtin_amdgcn_sched_group_barrier(SG_VMRD, 10, 0);
        __builtin_amdgcn_sched_group_barrier(SG_WMMA, 4, 0);
    }

    int lane = threadIdx.x & 31;
    int lg   = lane >> 4;
    int nn   = lane & 15;
    #pragma unroll
    for (int j = 0; j < 4; ++j) {
        int col = ng * 64 + j * 16 + nn;
        float bs = bias[col];
        #pragma unroll
        for (int r = 0; r < 8; ++r) {
            int t = row0 + r + 8 * lg;
            out[t * DD + col] = acc[j][r] + bs;
        }
    }
}

// ---------------- launch ----------------
extern "C" void kernel_launch(void* const* d_in, const int* in_sizes, int n_in,
                              void* d_out, int out_size, void* d_ws, size_t ws_size,
                              hipStream_t stream) {
    const float* x      = (const float*)d_in[0];
    const float* w_qkv  = (const float*)d_in[1];
    const float* b_qkv  = (const float*)d_in[2];
    const float* w_out  = (const float*)d_in[3];
    const float* b_out  = (const float*)d_in[4];
    float*       out    = (float*)d_out;

    char* ws = (char*)d_ws;
    _Float16* xh    = (_Float16*)ws;  ws += (size_t)MROWS * DD * 2;        // 6.3 MB
    _Float16* wqkvt = (_Float16*)ws;  ws += (size_t)NQKV * DD * 2;         // 3.5 MB
    _Float16* wot   = (_Float16*)ws;  ws += (size_t)DD * DD * 2;           // 1.2 MB
    _Float16* qk    = (_Float16*)ws;  ws += (size_t)MROWS * 2 * DD * 2;    // 12.6 MB
    _Float16* vt    = (_Float16*)ws;  ws += (size_t)BB * HH * HD * TT * 2; // 6.3 MB
    _Float16* yh    = (_Float16*)ws;                                       // 6.3 MB

    // 1) convert / transpose inputs to f16
    {
        int n = MROWS * DD;
        cvt_f32_to_f16<<<(n + 255) / 256, 256, 0, stream>>>(x, xh, n);
        n = DD * NQKV;
        transpose_f32_to_f16<<<(n + 255) / 256, 256, 0, stream>>>(w_qkv, wqkvt, DD, NQKV);
        n = DD * DD;
        transpose_f32_to_f16<<<(n + 255) / 256, 256, 0, stream>>>(w_out, wot, DD, DD);
    }

    // 2) QKV projection: 9216 waves (256 M-tiles x 36 N-groups), 4 waves/block
    gemm_qkv_kernel<<<(MROWS / 16) * (NQKV / 64) / 4, 128, 0, stream>>>(
        xh, wqkvt, b_qkv, qk, vt);

    // 3) flash attention: 3072 waves (2*12*128 q-tiles), 4 waves/block
    flash_attn_kernel<<<BB * HH * (TT / 16) / 4, 128, 0, stream>>>(qk, vt, yh);

    // 4) output projection: 3072 waves (256 M-tiles x 12 N-groups)
    gemm_out_kernel<<<(MROWS / 16) * (DD / 64) / 4, 128, 0, stream>>>(
        yh, wot, b_out, out);
}
